// S4D_model_24292335026644
// MI455X (gfx1250) — compile-verified, matrix operand order checked by reading, not run on hardware
//
#include <hip/hip_runtime.h>
#include <hip/hip_bf16.h>
#include <math.h>

// S4D forward for MI455X (gfx1250, wave32, WMMA, async global->LDS).
// Pipeline:
//   K1 setup   : per-(h,n) discretization -> tables Kt (conv kernel taps),
//                M1 (chunk-end-state matrix), G (inter-chunk output matrix), wT=w^64
//   K2 transpose: u (B,L,H) f32 -> uT (H, col=b*64+chunk, t) f16
//   K3 GEMM    : X[h] (64x512) = M1_h (64x64) @ uT_h (64x512)   [WMMA f16->f32]
//   K4 scan    : in-place chunk-carry scan on X (64 steps, 262144 lanes)
//   K5 GEMM    : Yt[h] = Toeplitz(Kt_h) @ uT_h + G_h @ X_h      [WMMA, fused, in-place]
//   K6 epilogue: Yt (H-major) -> out (B,L,H) + D*u + exact GELU (LDS tile transpose)

typedef __attribute__((ext_vector_type(16))) _Float16 v16h;
typedef __attribute__((ext_vector_type(8)))  float    v8f;

constexpr int Bsz = 8;
constexpr int Lsz = 4096;
constexpr int Hsz = 1024;
constexpr int Nst = 32;
constexpr int T   = 64;            // chunk length
constexpr int NC  = Lsz / T;       // 64 chunks
constexpr int COLS = Bsz * NC;     // 512 GEMM columns per h
constexpr int CW  = 64;            // columns per GEMM workgroup
constexpr int PAD = 72;            // LDS row pitch in halves (16B-aligned rows)

#if defined(__gfx1250__) &&                                                    \
    __has_builtin(__builtin_amdgcn_global_load_async_to_lds_b128) &&           \
    __has_builtin(__builtin_amdgcn_s_wait_asynccnt)
#define S4D_ASYNC 1
#else
#define S4D_ASYNC 0
#endif

#if S4D_ASYNC
// ROCm 7.2 prototype (from diagnostics): param 0 is AS1 pointer to int4-vector,
// param 1 the LDS-side pointer, then two constant ints (offset, cpol).
typedef __attribute__((__vector_size__(16))) int vi4;
typedef __attribute__((address_space(1))) vi4 g_vi4;
typedef __attribute__((address_space(3))) vi4 l_vi4;
__device__ inline void async_b128(const void* gsrc, void* ldst) {
  __builtin_amdgcn_global_load_async_to_lds_b128((g_vi4*)gsrc, (l_vi4*)ldst, 0,
                                                 0);
}
#endif

// ---------------------------------------------------------------------------
// WMMA fragment gather from LDS (rows padded to PAD halves = PAD/2 dwords).
// 16-bit A(16x32) per-lane layout (ISA 7.12.2): lane l: m=l&15, khalf=l>>4,
// K pairs -> dwords: j<4 -> khalf*4+j (K 0..15), j>=4 -> 8+khalf*4+(j-4) (K 16..31).
// B(32x16) uses the same map on B^T rows (n = l&15).
__device__ inline v16h gather16(const unsigned int* row, int kh, int kb16) {
  union { unsigned int u[8]; v16h h; } f;
#pragma unroll
  for (int j = 0; j < 8; ++j)
    f.u[j] = row[kb16 + kh * 4 + ((j < 4) ? j : (j + 4))];
  return f.h;
}

__device__ inline float gelu_exact(float x) {
  return 0.5f * x * (1.0f + erff(x * 0.70710678118654752440f));
}

// Stage a 64x64 f16 matrix (contiguous rows of 64 halves) into LDS rows of PAD.
__device__ inline void stage64x64(const _Float16* __restrict__ g,
                                  _Float16 (*l)[PAD], int tid) {
#if S4D_ASYNC
  for (int i = tid; i < 512; i += 128) {        // 16B chunks
    int r = i >> 3, q = i & 7;
    async_b128(g + r * 64 + q * 8, &l[r][q * 8]);
  }
#else
  for (int i = tid; i < 2048; i += 128) {       // dword chunks
    int r = i >> 5, d = i & 31;
    ((unsigned int*)&l[r][0])[d] = ((const unsigned int*)(g + r * 64))[d];
  }
#endif
}

// Stage CW rows of uT (row stride T halves in global) into LDS.
__device__ inline void stageB(const _Float16* __restrict__ uT, size_t rowbase,
                              _Float16 (*l)[PAD], int tid) {
#if S4D_ASYNC
  for (int i = tid; i < 512; i += 128) {
    int r = i >> 3, q = i & 7;
    async_b128(uT + (rowbase + r) * T + q * 8, &l[r][q * 8]);
  }
#else
  for (int i = tid; i < 2048; i += 128) {
    int r = i >> 5, d = i & 31;
    ((unsigned int*)&l[r][0])[d] =
        ((const unsigned int*)(uT + (rowbase + r) * T))[d];
  }
#endif
}

__device__ inline void stage_fence() {
#if S4D_ASYNC
  __builtin_amdgcn_s_wait_asynccnt(0);
#endif
  __syncthreads();
}

// ---------------------------------------------------------------------------
// K1: per-h tables.  block = 64 threads, grid = H.
__global__ __launch_bounds__(64) void k_setup(
    const float* __restrict__ log_dt, const float* __restrict__ A_real,
    const float* __restrict__ A_imag, const float* __restrict__ C_real,
    const float* __restrict__ C_imag, float* __restrict__ Kt,
    _Float16* __restrict__ M1, _Float16* __restrict__ G,
    float* __restrict__ wT) {
  const int h = blockIdx.x;
  const int t = threadIdx.x;
  __shared__ float s_lr[Nst], s_th[Nst], s_cr[Nst], s_ci[Nst];
  if (t < Nst) {
    float dt = expf(log_dt[h]);
    float ar = -expf(A_real[h * Nst + t]);
    float ai = A_imag[h * Nst + t];
    float lr = dt * ar, th = dt * ai;               // dtA = lr + i*th
    float er = expf(lr);
    float wr = er * cosf(th), wi = er * sinf(th);   // w = exp(dtA)
    float den = ar * ar + ai * ai;
    float br = ((wr - 1.f) * ar + wi * ai) / den;   // (w-1)/A
    float bi = (wi * ar - (wr - 1.f) * ai) / den;
    float Cr = C_real[h * Nst + t], Ci = C_imag[h * Nst + t];
    s_cr[t] = Cr * br - Ci * bi;                    // C' = C*(w-1)/A
    s_ci[t] = Cr * bi + Ci * br;
    s_lr[t] = lr; s_th[t] = th;
    float e64 = expf(64.f * lr);                    // wT = w^T
    wT[(size_t)(h * Nst + t) * 2 + 0] = e64 * cosf(64.f * th);
    wT[(size_t)(h * Nst + t) * 2 + 1] = e64 * sinf(64.f * th);
  }
  __syncthreads();
  { // Kt[h][d]: conv taps K[d] = 2*Re(sum_n C'_n w_n^d), d = t
    float acc = 0.f, d = (float)t;
    for (int n = 0; n < Nst; ++n) {
      float mag = expf(d * s_lr[n]);
      float ph = d * s_th[n];
      acc += s_cr[n] * (mag * cosf(ph)) - s_ci[n] * (mag * sinf(ph));
    }
    Kt[(size_t)h * T + t] = 2.f * acc;
  }
  { // M1[h][r][s]: r<32 -> Re(w_r^(63-s)), r>=32 -> Im(w_{r-32}^(63-s))
    int n = t & 31;
    float lr = s_lr[n], th = s_th[n];
    for (int s = 0; s < T; ++s) {
      float p = (float)(63 - s);
      float mag = expf(p * lr), ph = p * th;
      float val = (t < 32) ? mag * cosf(ph) : mag * sinf(ph);
      M1[(size_t)h * 4096 + t * T + s] = (_Float16)val;
    }
  }
  { // G[h][t][r]: y_inter[t] = sum_r G[t][r]*carry[r]; Z = C'_n w_n^(t+1)
    float p = (float)(t + 1);
    for (int r = 0; r < 2 * Nst; ++r) {
      int n = r & 31;
      float mag = expf(p * s_lr[n]), ph = p * s_th[n];
      float pr = mag * cosf(ph), pi = mag * sinf(ph);
      float Zr = s_cr[n] * pr - s_ci[n] * pi;
      float Zi = s_cr[n] * pi + s_ci[n] * pr;
      G[(size_t)h * 4096 + t * 64 + r] =
          (_Float16)((r < 32) ? (2.f * Zr) : (-2.f * Zi));
    }
  }
}

// ---------------------------------------------------------------------------
// K2: u (B,L,H) f32 -> uT[h][col][t] f16.  block(64,4), grid(H/64, B*L/64).
__global__ __launch_bounds__(256) void k_transpose_u(
    const float* __restrict__ u, _Float16* __restrict__ uT) {
  const int h0 = blockIdx.x * 64;
  const int bl0 = blockIdx.y * 64;       // tile of 64 consecutive l (one chunk)
  const int tx = threadIdx.x, ty = threadIdx.y;
  __shared__ float tile[64][65];
  for (int rr = ty; rr < 64; rr += 4)
    tile[rr][tx] = u[(size_t)(bl0 + rr) * Hsz + h0 + tx];
  __syncthreads();
  const int b = bl0 >> 12;               // / L
  const int l0 = bl0 & (Lsz - 1);
  const int col = b * NC + (l0 >> 6);
  for (int rr = ty; rr < 64; rr += 4)
    uT[((size_t)(h0 + rr) * COLS + col) * T + tx] = (_Float16)tile[tx][rr];
}

// ---------------------------------------------------------------------------
// K3: X[h] (64 r x 512 col) = M1_h @ uT_h.
// block=128 (4 waves), grid(COLS/CW=8, H).  Each wave: 4 column tiles, 8 WMMAs.
__global__ __launch_bounds__(128) void k_gemm_state(
    const _Float16* __restrict__ M1, const _Float16* __restrict__ uT,
    float* __restrict__ X) {
  const int h = blockIdx.y;
  const int col0 = blockIdx.x * CW;
  const int tid = threadIdx.x;
  __shared__ _Float16 lA[64][PAD];
  __shared__ _Float16 lB[CW][PAD];
  stage64x64(M1 + (size_t)h * 4096, lA, tid);
  stageB(uT, (size_t)h * COLS + col0, lB, tid);
  stage_fence();
  const int lane = tid & 31, wv = tid >> 5;
  const int m0 = wv * 16, kh = lane >> 4, nn = lane & 15;
  const unsigned int* arow = (const unsigned int*)&lA[m0 + nn][0];
  v8f acc[4];
#pragma unroll
  for (int ct = 0; ct < 4; ++ct)
    acc[ct] = v8f{0.f, 0.f, 0.f, 0.f, 0.f, 0.f, 0.f, 0.f};
#pragma unroll
  for (int kb16 = 0; kb16 <= 16; kb16 += 16) {
    v16h a = gather16(arow, kh, kb16);
#pragma unroll
    for (int ct = 0; ct < 4; ++ct) {
      v16h b = gather16((const unsigned int*)&lB[ct * 16 + nn][0], kh, kb16);
      acc[ct] = __builtin_amdgcn_wmma_f32_16x16x32_f16(false, a, false, b,
                                                       (short)0, acc[ct],
                                                       false, false);
    }
  }
#pragma unroll
  for (int ct = 0; ct < 4; ++ct) {
    float* o = X + ((size_t)(h * COLS + col0 + ct * 16 + nn)) * T + m0 + kh * 8;
#pragma unroll
    for (int v = 0; v < 8; ++v) o[v] = acc[ct][v];
  }
}

// ---------------------------------------------------------------------------
// K4: in-place chunk scan: carry[c] = wT*carry[c-1] + end[c-1]; slot c := carry[c].
// One lane per (h,b,n).  grid=1024, block=256.
__global__ __launch_bounds__(256) void k_scan(float* __restrict__ X,
                                              const float* __restrict__ wT) {
  const int gid = blockIdx.x * 256 + threadIdx.x;
  const int n = gid & 31;
  const int b = (gid >> 5) & 7;
  const int h = gid >> 8;
  const float wr = wT[(size_t)(h * Nst + n) * 2 + 0];
  const float wi = wT[(size_t)(h * Nst + n) * 2 + 1];
  float cr = 0.f, ci = 0.f;
  const size_t base = ((size_t)h * COLS + b * NC) * T;
  for (int c = 0; c < NC; ++c) {
    const size_t idx = base + (size_t)c * T;
    float er = X[idx + n];
    float ei = X[idx + 32 + n];
    X[idx + n] = cr;
    X[idx + 32 + n] = ci;
    float nr = wr * cr - wi * ci + er;
    float ni = wr * ci + wi * cr + ei;
    cr = nr; ci = ni;
  }
}

// ---------------------------------------------------------------------------
// K5: Yt[h] = Toeplitz(Kt_h) @ uT_h + G_h @ carry_h, written in place over X.
// block=128 (4 waves), grid(8, H).  Each wave: 4 column tiles, 16 WMMAs.
__global__ __launch_bounds__(128) void k_gemm_y(
    const float* __restrict__ Kt, const _Float16* __restrict__ G,
    const _Float16* __restrict__ uT, float* __restrict__ X) {
  const int h = blockIdx.y;
  const int col0 = blockIdx.x * CW;
  const int tid = threadIdx.x;
  __shared__ _Float16 lAk[64][PAD];   // Toeplitz of conv taps
  __shared__ _Float16 lAg[64][PAD];   // G matrix
  __shared__ _Float16 lBu[CW][PAD];   // u chunk columns
  __shared__ _Float16 lBx[CW][PAD];   // carry states (f32 -> f16)
  stage64x64(G + (size_t)h * 4096, lAg, tid);
  stageB(uT, (size_t)h * COLS + col0, lBu, tid);
  const float* kt = Kt + (size_t)h * T;
  for (int i = tid; i < 4096; i += 128) {
    int t = i >> 6, s = i & 63, d = t - s;
    lAk[t][s] = (d >= 0) ? (_Float16)kt[d] : (_Float16)0.f;
  }
  for (int i = tid; i < 4096; i += 128) {
    int cc = i >> 6, r = i & 63;
    lBx[cc][r] = (_Float16)X[((size_t)(h * COLS + col0 + cc)) * T + r];
  }
  stage_fence();
  const int lane = tid & 31, wv = tid >> 5;
  const int m0 = wv * 16, kh = lane >> 4, nn = lane & 15;
  const unsigned int* akrow = (const unsigned int*)&lAk[m0 + nn][0];
  const unsigned int* agrow = (const unsigned int*)&lAg[m0 + nn][0];
  v8f acc[4];
#pragma unroll
  for (int ct = 0; ct < 4; ++ct)
    acc[ct] = v8f{0.f, 0.f, 0.f, 0.f, 0.f, 0.f, 0.f, 0.f};
#pragma unroll
  for (int kb16 = 0; kb16 <= 16; kb16 += 16) {
    v16h ak = gather16(akrow, kh, kb16);
    v16h ag = gather16(agrow, kh, kb16);
#pragma unroll
    for (int ct = 0; ct < 4; ++ct) {
      v16h bu = gather16((const unsigned int*)&lBu[ct * 16 + nn][0], kh, kb16);
      acc[ct] = __builtin_amdgcn_wmma_f32_16x16x32_f16(false, ak, false, bu,
                                                       (short)0, acc[ct],
                                                       false, false);
      v16h bx = gather16((const unsigned int*)&lBx[ct * 16 + nn][0], kh, kb16);
      acc[ct] = __builtin_amdgcn_wmma_f32_16x16x32_f16(false, ag, false, bx,
                                                       (short)0, acc[ct],
                                                       false, false);
    }
  }
#pragma unroll
  for (int ct = 0; ct < 4; ++ct) {
    float* o = X + ((size_t)(h * COLS + col0 + ct * 16 + nn)) * T + m0 + kh * 8;
#pragma unroll
    for (int v = 0; v < 8; ++v) o[v] = acc[ct][v];
  }
}

// ---------------------------------------------------------------------------
// K6: Yt[h][col][t] -> out[b][l][h] with y + D*u and exact GELU.
__global__ __launch_bounds__(256) void k_epilogue(
    const float* __restrict__ Yt, const float* __restrict__ u,
    const float* __restrict__ Dp, float* __restrict__ out) {
  const int h0 = blockIdx.x * 64;
  const int bl0 = blockIdx.y * 64;
  const int tx = threadIdx.x, ty = threadIdx.y;
  const int b = bl0 >> 12;
  const int l0 = bl0 & (Lsz - 1);
  const int col = b * NC + (l0 >> 6);
  __shared__ float tile[64][65];
  for (int rr = ty; rr < 64; rr += 4)
    tile[rr][tx] = Yt[((size_t)(h0 + rr) * COLS + col) * T + tx];
  __syncthreads();
  const float D = Dp[0];
  for (int rr = ty; rr < 64; rr += 4) {
    size_t gi = (size_t)(bl0 + rr) * Hsz + h0 + tx;
    float y = tile[tx][rr] + D * u[gi];
    out[gi] = gelu_exact(y);
  }
}

// ---------------------------------------------------------------------------
extern "C" void kernel_launch(void* const* d_in, const int* in_sizes, int n_in,
                              void* d_out, int out_size, void* d_ws,
                              size_t ws_size, hipStream_t stream) {
  const float* u      = (const float*)d_in[0];
  const float* log_dt = (const float*)d_in[1];
  const float* A_real = (const float*)d_in[2];
  const float* A_imag = (const float*)d_in[3];
  const float* C_real = (const float*)d_in[4];
  const float* C_imag = (const float*)d_in[5];
  const float* Dp     = (const float*)d_in[6];
  float* out = (float*)d_out;

  // Workspace layout (all offsets 256B-aligned):
  char* ws = (char*)d_ws;
  _Float16* uT = (_Float16*)ws;                       //  67,108,864 B
  float*    X  = (float*)(ws + 67108864);             // 134,217,728 B
  _Float16* M1 = (_Float16*)(ws + 201326592);         //   8,388,608 B
  _Float16* G  = (_Float16*)(ws + 209715200);         //   8,388,608 B
  float*    Kt = (float*)(ws + 218103808);            //     262,144 B
  float*    wT = (float*)(ws + 218365952);            //     262,144 B
  // total 218,628,096 B

  k_setup<<<Hsz, 64, 0, stream>>>(log_dt, A_real, A_imag, C_real, C_imag, Kt,
                                  M1, G, wT);
  k_transpose_u<<<dim3(Hsz / 64, (Bsz * Lsz) / 64), dim3(64, 4), 0, stream>>>(
      u, uT);
  k_gemm_state<<<dim3(COLS / CW, Hsz), 128, 0, stream>>>(M1, uT, X);
  k_scan<<<(Hsz * Bsz * Nst) / 256, 256, 0, stream>>>(X, wT);
  k_gemm_y<<<dim3(COLS / CW, Hsz), 128, 0, stream>>>(Kt, G, uT, X);
  k_epilogue<<<dim3(Hsz / 64, (Bsz * Lsz) / 64), dim3(64, 4), 0, stream>>>(
      X, u, Dp, out);
}
